// MSDeformAttnPixelDecoder_31490700214567
// MI455X (gfx1250) — matrix operand order repaired and checked by reading, not compiled, not run on hardware
//
#include <hip/hip_runtime.h>
#include <hip/hip_bf16.h>
#include <math.h>

// ---------------------------------------------------------------------------
// MSDeformAttn pixel decoder for MI455X (gfx1250).
// All dense GEMMs (~190 GFLOP total) run on v_wmma_f32_16x16x32_bf16 with
// fp32 accumulation; gather/normalization stages are fp32 VALU kernels.
// ---------------------------------------------------------------------------

typedef __attribute__((ext_vector_type(16))) __bf16 v16bf;
typedef __attribute__((ext_vector_type(8)))  float  v8f;

constexpr int kB = 2, kD = 256, kHEADS = 4, kNLVL = 4, kPTS = 4;
constexpr int kLQ = 21760, kDH = 64, kFFN = 1024;
constexpr long kBLQ = (long)kB * kLQ;     // 43520

__constant__ int c_HW[4]   = {16, 32, 64, 128};
__constant__ int c_QOFF[5] = {0, 256, 1280, 5376, 21760};

// Pack two fp32 -> two bf16 -> one dword LDS store (v_cvt_pk_bf16_f32 + ds_store_b32).
__device__ inline void pack_store_bf16x2(__bf16* dst, float f0, float f1)
{
    union { __bf16 h[2]; unsigned int u; } p;
    p.h[0] = (__bf16)f0; p.h[1] = (__bf16)f1;
    *reinterpret_cast<unsigned int*>(dst) = p.u;
}

// ---------------------------------------------------------------------------
// Generic tiled GEMM: C(M,N) = A(M,K) @ B(K,N) + bias, optional ReLU.
// A[m,k] = Aptr[m*a_rs + k*a_cs], B[k,n] = Bptr[k*b_rs + n*b_cs].
// ACS1=1 specializes the common a_cs==1 case to vector (float2) A loads.
// 32-bit element offsets (largest buffer ~178 MB) -> saddr + voffset form.
// Software pipeline: global loads for tile k+1 are issued right after the
// barrier, so they are in flight while tile k's ds_load_b128 fragments and
// WMMAs execute (single loadcnt wait per tile, full MLP).
// Block: 256 threads = 8 wave32; block tile 128x64, wave grid 4x2, each wave
// a 32x32 tile (2x2 WMMA accumulators). Requires M%128==0, N%64==0, K%32==0
// (true for every GEMM in this model).
// ---------------------------------------------------------------------------
template<int RELU, int ACS1>
__global__ __launch_bounds__(256)
void gemm_bf16_wmma(const float* __restrict__ A, int a_rs, int a_cs,
                    const float* __restrict__ Bm, int b_rs, int b_cs,
                    const float* __restrict__ bias,
                    float* __restrict__ C, int c_rs,
                    int K)
{
    constexpr int BM = 128, BN = 64, BK = 32, PAD = 8;   // PAD even (dword align)
    __shared__ __bf16 As[BM][BK + PAD];
    __shared__ __bf16 Bs[BN][BK + PAD];   // Bs[n][k]

    const int row0 = blockIdx.y * BM;
    const int col0 = blockIdx.x * BN;
    const int tid  = threadIdx.x;
    const int lane = tid & 31;
    const int wave = tid >> 5;
    const int wr = wave >> 1, wc = wave & 1;             // 4 x 2 wave grid

    v8f acc[2][2];
    for (int i = 0; i < 2; i++)
        for (int j = 0; j < 2; j++)
            for (int e = 0; e < 8; e++) acc[i][j][e] = 0.0f;

    float2 aT[8], bT[4];   // register-staged K-tile (A: 8 pairs, B: 4 pairs)

    auto load_a = [&](int k0, float2* t) {
        for (int i = 0; i < 8; i++) {
            int p  = i * 256 + tid;
            int r  = p >> 4;
            int kp = (p & 15) * 2;
            if (ACS1) {
                unsigned off = (unsigned)(row0 + r) * (unsigned)a_rs + (unsigned)(k0 + kp);
                t[i] = *reinterpret_cast<const float2*>(A + off);
            } else {
                unsigned off = (unsigned)(row0 + r) * (unsigned)a_rs
                             + (unsigned)(k0 + kp) * (unsigned)a_cs;
                t[i].x = A[off];
                t[i].y = A[off + (unsigned)a_cs];
            }
        }
    };
    auto load_b = [&](int k0, float2* t) {
        for (int i = 0; i < 4; i++) {
            int p  = i * 256 + tid;
            int n  = p >> 4;
            int kp = (p & 15) * 2;
            unsigned off = (unsigned)(k0 + kp) * (unsigned)b_rs
                         + (unsigned)(col0 + n) * (unsigned)b_cs;
            t[i].x = Bm[off];
            t[i].y = Bm[off + (unsigned)b_rs];
        }
    };

    load_a(0, aT);
    load_b(0, bT);

    for (int k0 = 0; k0 < K; k0 += BK) {
        // ---- Drain register buffer into LDS (convert fp32 -> bf16 pairs).
        for (int i = 0; i < 8; i++) {
            int p = i * 256 + tid;
            pack_store_bf16x2(&As[p >> 4][(p & 15) * 2], aT[i].x, aT[i].y);
        }
        for (int i = 0; i < 4; i++) {
            int p = i * 256 + tid;
            pack_store_bf16x2(&Bs[p >> 4][(p & 15) * 2], bT[i].x, bT[i].y);
        }
        __syncthreads();

        // ---- Kick off next tile's global loads (hidden behind WMMAs below).
        if (k0 + BK < K) {
            load_a(k0 + BK, aT);
            load_b(k0 + BK, bT);
        }

        // ---- Build fragments per ISA 7.12.2 16-bit layouts.
        v16bf afrag[2], bfrag[2];
        for (int tm = 0; tm < 2; tm++) {
            int r  = wr * 32 + tm * 16 + (lane & 15);
            int kb = (lane < 16) ? 0 : 8;
            for (int e = 0; e < 8; e++) afrag[tm][e]     = As[r][kb + e];
            for (int e = 0; e < 8; e++) afrag[tm][8 + e] = As[r][16 + kb + e];
        }
        for (int tn = 0; tn < 2; tn++) {
            int n  = wc * 32 + tn * 16 + (lane & 15);
            int kb = (lane < 16) ? 0 : 8;
            for (int e = 0; e < 8; e++) bfrag[tn][e]     = Bs[n][kb + e];
            for (int e = 0; e < 8; e++) bfrag[tn][8 + e] = Bs[n][16 + kb + e];
        }

        for (int tm = 0; tm < 2; tm++)
            for (int tn = 0; tn < 2; tn++)
                acc[tm][tn] = __builtin_amdgcn_wmma_f32_16x16x32_bf16(
                    false, afrag[tm], false, bfrag[tn],
                    (short)0, acc[tm][tn], false, false);
        __syncthreads();
    }

    // Epilogue: C layout lane<16 -> M=v, lane>=16 -> M=8+v; N = lane%16.
    for (int tm = 0; tm < 2; tm++)
        for (int tn = 0; tn < 2; tn++) {
            int col  = col0 + wc * 32 + tn * 16 + (lane & 15);
            float bv = bias[col];
            for (int v = 0; v < 8; v++) {
                int r   = row0 + wr * 32 + tm * 16 + v + ((lane < 16) ? 0 : 8);
                float x = acc[tm][tn][v] + bv;
                if (RELU) x = fmaxf(x, 0.0f);
                C[(unsigned)r * (unsigned)c_rs + (unsigned)col] = x;
            }
        }
}

// ---------------------------------------------------------------------------
// GroupNorm over (8 channels x HW) per (batch, group). grid = B*32.
// ---------------------------------------------------------------------------
__global__ __launch_bounds__(256)
void groupnorm_kernel(float* __restrict__ src, const float* __restrict__ gg,
                      const float* __restrict__ gb, int q0, int n)
{
    __shared__ float s_sum[256], s_sq[256];
    int b = blockIdx.x >> 5;
    int g = blockIdx.x & 31;
    long base = ((long)b * kLQ + q0) * kD + g * 8;
    int total = n * 8;
    float sum = 0.f, sq = 0.f;
    for (int i = threadIdx.x; i < total; i += 256) {
        float v = src[base + (long)(i >> 3) * kD + (i & 7)];
        sum += v; sq += v * v;
    }
    s_sum[threadIdx.x] = sum; s_sq[threadIdx.x] = sq;
    __syncthreads();
    for (int s = 128; s > 0; s >>= 1) {
        if (threadIdx.x < s) {
            s_sum[threadIdx.x] += s_sum[threadIdx.x + s];
            s_sq[threadIdx.x]  += s_sq[threadIdx.x + s];
        }
        __syncthreads();
    }
    float mean = s_sum[0] / total;
    float var  = s_sq[0] / total - mean * mean;
    float inv  = rsqrtf(var + 1e-5f);
    for (int i = threadIdx.x; i < total; i += 256) {
        int j = i & 7;
        long idx = base + (long)(i >> 3) * kD + j;
        int c = g * 8 + j;
        src[idx] = (src[idx] - mean) * inv * gg[c] + gb[c];
    }
}

// ---------------------------------------------------------------------------
// Sine positional embedding + level embedding. One thread per (q,d).
// ---------------------------------------------------------------------------
__global__ void pos_kernel(float* __restrict__ pos,
                           const float* __restrict__ level_embed)
{
    int idx = blockIdx.x * 256 + threadIdx.x;
    if (idx >= kLQ * kD) return;
    int q = idx / kD, d = idx % kD;
    int lv = 0;
    while (lv < 3 && q >= c_QOFF[lv + 1]) lv++;
    int HW = c_HW[lv];
    int qi = q - c_QOFF[lv];
    int y = qi / HW, x = qi % HW;
    const float scale = 6.28318530717958647692f;
    int i; float e;
    if (d < 128) { i = d;       e = (float)(y + 1) / ((float)HW + 1e-6f) * scale; }
    else         { i = d - 128; e = (float)(x + 1) / ((float)HW + 1e-6f) * scale; }
    float t = powf(10000.0f, (float)(2 * (i / 2)) / 128.0f);
    float p = e / t;
    float v = (i & 1) ? cosf(p) : sinf(p);
    pos[idx] = v + level_embed[lv * kD + d];
}

// q = src + pos (pos broadcast across batch).
__global__ void addq_kernel(float* __restrict__ q, const float* __restrict__ src,
                            const float* __restrict__ pos)
{
    long idx = (long)blockIdx.x * 256 + threadIdx.x;
    long p = idx;
    if (p >= (long)kLQ * kD) p -= (long)kLQ * kD;   // B == 2
    q[idx] = src[idx] + pos[p];
}

// Softmax over the 16 contiguous (level,point) logits per (b,q,head).
__global__ void softmax16_kernel(float* __restrict__ aw)
{
    int t = blockIdx.x * 256 + threadIdx.x;
    if (t >= (int)(kBLQ * kHEADS)) return;
    long base = (long)t * 16;
    float v[16], m = -1e30f;
    for (int i = 0; i < 16; i++) { v[i] = aw[base + i]; m = fmaxf(m, v[i]); }
    float s = 0.f;
    for (int i = 0; i < 16; i++) { v[i] = expf(v[i] - m); s += v[i]; }
    float inv = 1.0f / s;
    for (int i = 0; i < 16; i++) aw[base + i] = v[i] * inv;
}

// ---------------------------------------------------------------------------
// Deformable bilinear sampling. One wave per (b,q,head); each lane owns 2 of
// the 64 head channels. grid_sample align_corners=False, zero padding.
// ---------------------------------------------------------------------------
__global__ __launch_bounds__(256)
void sample_kernel(const float* __restrict__ val, const float* __restrict__ off,
                   const float* __restrict__ aw, float* __restrict__ samp)
{
    int gid  = blockIdx.x * 8 + (threadIdx.x >> 5);
    int lane = threadIdx.x & 31;
    if (gid >= (int)(kBLQ * kHEADS)) return;
    int b = gid / (kLQ * kHEADS);
    int rem = gid % (kLQ * kHEADS);
    int q = rem / kHEADS;
    int h = rem % kHEADS;

    int qlv = 0;
    while (qlv < 3 && q >= c_QOFF[qlv + 1]) qlv++;
    int Wq = c_HW[qlv];
    int qi = q - c_QOFF[qlv];
    float refx = ((qi % Wq) + 0.5f) / (float)Wq;
    float refy = ((qi / Wq) + 0.5f) / (float)Wq;   // H == W for every level

    int dh0 = lane * 2;
    float acc0 = 0.f, acc1 = 0.f;
    long offbase = ((long)b * kLQ + q) * 128 + h * 32;
    long awbase  = ((long)b * kLQ + q) * 64  + h * 16;

    for (int lv = 0; lv < 4; lv++) {
        int L = c_HW[lv];
        long vbase = ((long)b * kLQ + c_QOFF[lv]) * kD + h * kDH + dh0;
        for (int pt = 0; pt < 4; pt++) {
            float ox = off[offbase + (lv * 4 + pt) * 2 + 0];
            float oy = off[offbase + (lv * 4 + pt) * 2 + 1];
            float w  = aw[awbase + lv * 4 + pt];
            float xx = (refx + ox / (float)L) * L - 0.5f;
            float yy = (refy + oy / (float)L) * L - 0.5f;
            float xf = floorf(xx), yf = floorf(yy);
            float tx = xx - xf, ty = yy - yf;
            int x0 = (int)xf, y0 = (int)yf;
            float wt[4] = {(1 - ty) * (1 - tx), (1 - ty) * tx,
                           ty * (1 - tx),       ty * tx};
            for (int c = 0; c < 4; c++) {
                int xi = x0 + (c & 1), yi = y0 + (c >> 1);
                if (xi < 0 || xi >= L || yi < 0 || yi >= L) continue;
                const float2 v2 =
                    *reinterpret_cast<const float2*>(val + vbase + (long)(yi * L + xi) * kD);
                float ww = w * wt[c];
                acc0 += ww * v2.x; acc1 += ww * v2.y;
            }
        }
    }
    long obase = ((long)b * kLQ + q) * kD + h * kDH + dh0;
    samp[obase] = acc0; samp[obase + 1] = acc1;
}

// LayerNorm of (src + attn_out) over D=256. One wave per row.
__global__ __launch_bounds__(256)
void layernorm_kernel(float* __restrict__ src, const float* __restrict__ aout,
                      const float* __restrict__ g, const float* __restrict__ bta)
{
    int row  = blockIdx.x * 8 + (threadIdx.x >> 5);
    int lane = threadIdx.x & 31;
    long base = (long)row * kD;
    float x[8], sum = 0.f;
    for (int i = 0; i < 8; i++) {
        x[i] = src[base + lane + i * 32] + aout[base + lane + i * 32];
        sum += x[i];
    }
    for (int o = 16; o > 0; o >>= 1) sum += __shfl_xor(sum, o, 32);
    float mean = sum / kD;
    float sq = 0.f;
    for (int i = 0; i < 8; i++) { float d = x[i] - mean; sq += d * d; }
    for (int o = 16; o > 0; o >>= 1) sq += __shfl_xor(sq, o, 32);
    float inv = rsqrtf(sq / kD + 1e-5f);
    for (int i = 0; i < 8; i++) {
        int d = lane + i * 32;
        src[base + d] = (x[i] - mean) * inv * g[d] + bta[d];
    }
}

// ---------------------------------------------------------------------------
// Host orchestration.
// ---------------------------------------------------------------------------
extern "C" void kernel_launch(void* const* d_in, const int* in_sizes, int n_in,
                              void* d_out, int out_size, void* d_ws, size_t ws_size,
                              hipStream_t stream)
{
    (void)in_sizes; (void)n_in; (void)out_size; (void)ws_size;

    const float *x[4], *pw[4], *pb[4], *gg[4], *gb[4];
    for (int i = 0; i < 4; i++) {
        x[i]  = (const float*)d_in[i * 5 + 0];
        pw[i] = (const float*)d_in[i * 5 + 1];
        pb[i] = (const float*)d_in[i * 5 + 2];
        gg[i] = (const float*)d_in[i * 5 + 3];
        gb[i] = (const float*)d_in[i * 5 + 4];
    }
    const float* level_embed = (const float*)d_in[20];
    const float* Woff  = (const float*)d_in[21];
    const float* boff  = (const float*)d_in[22];
    const float* Wattn = (const float*)d_in[23];
    const float* battn = (const float*)d_in[24];
    const float* Wval  = (const float*)d_in[25];
    const float* bval  = (const float*)d_in[26];
    const float* Wout  = (const float*)d_in[27];
    const float* bout  = (const float*)d_in[28];
    const float* ng    = (const float*)d_in[29];
    const float* nb    = (const float*)d_in[30];
    const float* W1    = (const float*)d_in[31];
    const float* b1    = (const float*)d_in[32];
    const float* W2    = (const float*)d_in[33];
    const float* b2    = (const float*)d_in[34];

    // Workspace layout (floats). hid aliases the dead attention scratch.
    constexpr long SZ_SRC = kBLQ * kD;                       // 11,141,120
    constexpr long OFF_SRC  = 0;
    constexpr long OFF_POS  = OFF_SRC + SZ_SRC;
    constexpr long OFF_AOUT = OFF_POS + (long)kLQ * kD;
    constexpr long OFF_Q    = OFF_AOUT + SZ_SRC;
    constexpr long OFF_OFF  = OFF_Q + SZ_SRC;
    constexpr long OFF_AW   = OFF_OFF + kBLQ * 128;
    constexpr long OFF_VAL  = OFF_AW + kBLQ * 64;
    constexpr long OFF_SAMP = OFF_VAL + SZ_SRC;
    constexpr long OFF_HID  = OFF_Q;                         // alias (hid live only after LN)

    float* ws    = (float*)d_ws;
    float* src   = ws + OFF_SRC;
    float* pos   = ws + OFF_POS;
    float* aout  = ws + OFF_AOUT;
    float* qbuf  = ws + OFF_Q;
    float* offb  = ws + OFF_OFF;
    float* awb   = ws + OFF_AW;
    float* valb  = ws + OFF_VAL;
    float* sampb = ws + OFF_SAMP;
    float* hidb  = ws + OFF_HID;

    auto gemm = [&](const float* A, int ars, int acs,
                    const float* Bm, int brs, int bcs,
                    const float* bias, float* C, int crs,
                    int M, int N, int K, int relu) {
        dim3 grid(N / 64, M / 128);
        if (acs == 1) {
            if (relu)
                gemm_bf16_wmma<1, 1><<<grid, dim3(256), 0, stream>>>(A, ars, acs, Bm, brs, bcs,
                                                                     bias, C, crs, K);
            else
                gemm_bf16_wmma<0, 1><<<grid, dim3(256), 0, stream>>>(A, ars, acs, Bm, brs, bcs,
                                                                     bias, C, crs, K);
        } else {
            gemm_bf16_wmma<0, 0><<<grid, dim3(256), 0, stream>>>(A, ars, acs, Bm, brs, bcs,
                                                                 bias, C, crs, K);
        }
    };

    const int HWs[4]  = {256, 1024, 4096, 16384};
    const int CIN[4]  = {2048, 1024, 512, 256};
    const int QOFF[4] = {0, 256, 1280, 5376};

    // Stage 1: per-level 1x1 conv (as strided GEMM) + GroupNorm.
    for (int lv = 0; lv < 4; lv++) {
        for (int b = 0; b < kB; b++) {
            const float* Ab = x[lv] + (long)b * CIN[lv] * HWs[lv]; // A[q,c]=x[c*HW+q]
            float* Cb = src + ((long)b * kLQ + QOFF[lv]) * kD;
            gemm(Ab, 1, HWs[lv], pw[lv], 1, CIN[lv], pb[lv], Cb, kD,
                 HWs[lv], kD, CIN[lv], 0);
        }
        groupnorm_kernel<<<kB * 32, 256, 0, stream>>>(src, gg[lv], gb[lv],
                                                      QOFF[lv], HWs[lv]);
    }
    pos_kernel<<<(kLQ * kD) / 256, 256, 0, stream>>>(pos, level_embed);

    // Stage 2: 3 decoder layers.
    for (int l = 0; l < 3; l++) {
        const float* Woff_l  = Woff  + (long)l * kD * 128;
        const float* boff_l  = boff  + (long)l * 128;
        const float* Wattn_l = Wattn + (long)l * kD * 64;
        const float* battn_l = battn + (long)l * 64;
        const float* Wval_l  = Wval  + (long)l * kD * kD;
        const float* bval_l  = bval  + (long)l * kD;
        const float* Wout_l  = Wout  + (long)l * kD * kD;
        const float* bout_l  = bout  + (long)l * kD;
        const float* ng_l    = ng    + (long)l * kD;
        const float* nb_l    = nb    + (long)l * kD;
        const float* W1_l    = W1    + (long)l * kD * kFFN;
        const float* b1_l    = b1    + (long)l * kFFN;
        const float* W2_l    = W2    + (long)l * kFFN * kD;
        const float* b2_l    = b2    + (long)l * kD;

        addq_kernel<<<(int)(kBLQ * kD / 256), 256, 0, stream>>>(qbuf, src, pos);
        gemm(qbuf, kD, 1, Woff_l, 128, 1, boff_l, offb, 128, (int)kBLQ, 128, kD, 0);
        gemm(qbuf, kD, 1, Wattn_l, 64, 1, battn_l, awb, 64, (int)kBLQ, 64, kD, 0);
        softmax16_kernel<<<(int)((kBLQ * kHEADS + 255) / 256), 256, 0, stream>>>(awb);
        gemm(src, kD, 1, Wval_l, kD, 1, bval_l, valb, kD, (int)kBLQ, kD, kD, 0);
        sample_kernel<<<(int)(kBLQ * kHEADS / 8), 256, 0, stream>>>(valb, offb, awb, sampb);
        gemm(sampb, kD, 1, Wout_l, kD, 1, bout_l, aout, kD, (int)kBLQ, kD, kD, 0);
        layernorm_kernel<<<(int)(kBLQ / 8), 256, 0, stream>>>(src, aout, ng_l, nb_l);
        gemm(src, kD, 1, W1_l, kFFN, 1, b1_l, hidb, kFFN, (int)kBLQ, kFFN, kD, 1);
        float* dst = (l == 2) ? (float*)d_out : src;  // final MLP output is the result
        gemm(hidb, kFFN, 1, W2_l, kD, 1, b2_l, dst, kD, (int)kBLQ, kD, kFFN, 0);
    }
}
